// Model3_41034117546431
// MI455X (gfx1250) — compile-verified, matrix operand order checked by reading, not run on hardware
//
#include <hip/hip_runtime.h>
#include <hip/hip_bf16.h>
#include <math.h>

// ---------------------------------------------------------------------------
// MI455X / gfx1250 implementation. All dense layers run through
// v_wmma_f32_16x16x32_bf16 (bf16 in, f32 accumulate) with operands pre-packed
// into the wave32 WMMA fragment layout (one 32B contiguous slice per lane).
// Each GEMM wave computes a 32x32 macro-tile: 4 wmma per K-step with 2x
// operand reuse. The sequential BiLSTM recurrence keeps Whh resident in LDS
// as bf16 (180 KB of the WGP's 320 KB).
// ---------------------------------------------------------------------------

typedef __bf16 bf16_t;
typedef __attribute__((ext_vector_type(16))) __bf16 v16bf;
typedef __attribute__((ext_vector_type(8)))  float  v8f;

// ---- model dims (match reference) ----
#define D_     300
#define T_     8192
#define L_     16
#define CD_    30
#define F_     50
#define H_     150
#define G4_    600      // 4*H
#define E_     128
#define NC_    8
#define A_     16
#define C_     4
#define LME_   10
#define LMC_   64
#define FD_    20
#define SED_   920      // 3*D + FD
#define XIN_   450      // D + 3*F
#define XKP_   480      // pad(450 -> mult 32)
#define ZLD_   608      // pad(600 -> mult 32)
#define HKP_   160      // pad(150 -> mult 32)
#define DKP_   320      // pad(300 -> mult 32)
#define SEDKP_ 928      // pad(920 -> mult 32)
#define SMLD_  320      // pad(300 -> mult 32)
#define GLD_   32       // pad(30  -> mult 32)
#define RELKP_ 2784     // pad(2780 -> mult 32)

// ---- input indices (setup_inputs dict order; params in insertion order) ----
enum {
  IN_WORD=0, IN_CHAR=1, IN_ESPAN=2, IN_CB=3, IN_CF=4, IN_AA=5, IN_DF=6,
  IN_FT=7, IN_ACTOR=8, IN_CEMB=9,
  IN_CW0=10, IN_CB0=11, IN_CW1=12, IN_CB1=13, IN_CW2=14, IN_CB2=15,
  IN_LF_WIH=16, IN_LF_WHH=17, IN_LF_B=18,
  IN_LB_WIH=19, IN_LB_WHH=20, IN_LB_B=21,
  IN_EFEAT=22, IN_CFEAT=23, IN_DFEAT=24,
  IN_A0W=25, IN_A0B=26, IN_A1W=27, IN_A1B=28, IN_A2W=29, IN_A2B=30,
  IN_G0W=31, IN_G0B=32, IN_G1W=33, IN_G1B=34, IN_G2W=35, IN_G2B=36,
  IN_AFW=37, IN_AFB=38,
  IN_R0W=39, IN_R0B=40, IN_R1W=41, IN_R1B=42, IN_R2W=43, IN_R2B=44
};

// ---------------------------------------------------------------------------
// WMMA fragment swizzle offsets (bf16 element index in packed buffer).
// Fragment = one 16(M/N) x 32(K) tile; per lane a contiguous 16-element slice.
// A-matrix (16x32, ISA 7.12.2): lane = hl*16 + (m&15), elem j covers
//   K = hl*8 + j (j<8)  and  K = 16 + hl*8 + (j-8) (j>=8).
// B-matrix mirrors A with N in place of M.
// ---------------------------------------------------------------------------
__device__ __forceinline__ size_t asw_off(int m, int k, int tK) {
  int mt = m >> 4, kt = k >> 5, kin = k & 31;
  int hl = (kin >> 3) & 1;
  int j  = (kin & 7) | ((kin >> 4) << 3);
  int lane = (hl << 4) | (m & 15);
  return ((size_t)((size_t)mt * tK + kt) * 32 + lane) * 16 + j;
}
__device__ __forceinline__ size_t bsw_off(int k, int n, int tN) {
  int nt = n >> 4, kt = k >> 5, kin = k & 31;
  int hl = (kin >> 3) & 1;
  int j  = (kin & 7) | ((kin >> 4) << 3);
  int lane = (hl << 4) | (n & 15);
  return ((size_t)((size_t)kt * tN + nt) * 32 + lane) * 16 + j;
}

// ---------------------------------------------------------------------------
// Packing kernels (fp32 -> zero-padded, fragment-swizzled bf16)
// ---------------------------------------------------------------------------

// activations: src f32 [M x ld] -> swizzled A bf16 [M x Kp]
__global__ void k_pack_a(const float* __restrict__ src, int ld,
                         bf16_t* __restrict__ dst, int M, int K, int Kp) {
  long long idx = (long long)blockIdx.x * blockDim.x + threadIdx.x;
  long long total = (long long)M * Kp;
  if (idx >= total) return;
  int m = (int)(idx / Kp), k = (int)(idx % Kp);
  float v = (k < K) ? src[(size_t)m * ld + k] : 0.f;
  dst[asw_off(m, k, Kp >> 5)] = (bf16_t)v;
}

// weight used as x@W, W f32 [K x N] -> swizzled B bf16 [Kp x N16]
__global__ void k_pack_w(const float* __restrict__ W, bf16_t* __restrict__ dst,
                         int K, int N, int Kp, int N16) {
  long long idx = (long long)blockIdx.x * blockDim.x + threadIdx.x;
  long long total = (long long)Kp * N16;
  if (idx >= total) return;
  int k = (int)(idx / N16), n = (int)(idx % N16);
  float v = (k < K && n < N) ? W[(size_t)k * N + n] : 0.f;
  dst[bsw_off(k, n, N16 >> 4)] = (bf16_t)v;
}

// weight used as x@W^T (or B = M^T): W f32 [N x ldw] -> swizzled B bf16
__global__ void k_pack_wT(const float* __restrict__ W, int ldw,
                          bf16_t* __restrict__ dst, int K, int N, int Kp, int N16) {
  long long idx = (long long)blockIdx.x * blockDim.x + threadIdx.x;
  long long total = (long long)Kp * N16;
  if (idx >= total) return;
  int k = (int)(idx / N16), n = (int)(idx % N16);
  float v = (k < K && n < N) ? W[(size_t)n * ldw + k] : 0.f;
  dst[bsw_off(k, n, N16 >> 4)] = (bf16_t)v;
}

// ---------------------------------------------------------------------------
// WMMA bf16 GEMM on swizzled operands.
// C[M x N16](f32) = A[M x Kp](bf16) @ B[Kp x N16](bf16) + bias, opt. ReLU.
// One 32x32 macro-tile per wave32: 2 A-frags x 2 B-frags -> 4 wmma / K-step.
// M % 32 == 0, Kp % 32 == 0, N16 % 32 == 0. Stores only cols < Nreal.
// ---------------------------------------------------------------------------
__device__ __forceinline__ void store_tile(float* __restrict__ Cf, int N16, int Nreal,
                                           int relu, const float* __restrict__ bias,
                                           int mt, int nt, v8f acc, int lane) {
  int n = (nt << 4) + (lane & 15);
  if (n >= Nreal) return;
  int m0 = (mt << 4) + ((lane >> 4) << 3);
  float bv = bias ? bias[n] : 0.f;
#pragma unroll
  for (int rr = 0; rr < 8; ++rr) {
    float v = acc[rr] + bv;
    if (relu) v = fmaxf(v, 0.f);
    Cf[(size_t)(m0 + rr) * N16 + n] = v;
  }
}

__global__ void k_gemm(const bf16_t* __restrict__ Ab, const bf16_t* __restrict__ Bb,
                       const float* __restrict__ bias, float* __restrict__ Cf,
                       int M, int Nreal, int N16, int Kp, int relu) {
  const v16bf* Af = (const v16bf*)Ab;
  const v16bf* Bf = (const v16bf*)Bb;
  int gw   = (blockIdx.x * blockDim.x + threadIdx.x) >> 5;
  int lane = threadIdx.x & 31;
  int tN = N16 >> 4, tK = Kp >> 5;
  int tM2 = M >> 5, tN2 = N16 >> 5;
  if (gw >= tM2 * tN2) return;
  int mt0 = (gw / tN2) << 1;
  int nt0 = (gw % tN2) << 1;
  v8f acc00 = {0.f,0.f,0.f,0.f,0.f,0.f,0.f,0.f};
  v8f acc01 = acc00, acc10 = acc00, acc11 = acc00;
  const v16bf* pa0 = Af + ((size_t)mt0       * tK) * 32 + lane;
  const v16bf* pa1 = Af + ((size_t)(mt0 + 1) * tK) * 32 + lane;
  const v16bf* pb0 = Bf + ((size_t)nt0)       * 32 + lane;
  const v16bf* pb1 = Bf + ((size_t)(nt0 + 1)) * 32 + lane;
  for (int kt = 0; kt < tK; ++kt) {
    v16bf a0 = pa0[(size_t)kt * 32];
    v16bf a1 = pa1[(size_t)kt * 32];
    v16bf b0 = pb0[(size_t)kt * tN * 32];
    v16bf b1 = pb1[(size_t)kt * tN * 32];
    acc00 = __builtin_amdgcn_wmma_f32_16x16x32_bf16(false, a0, false, b0, (short)0, acc00, false, false);
    acc01 = __builtin_amdgcn_wmma_f32_16x16x32_bf16(false, a0, false, b1, (short)0, acc01, false, false);
    acc10 = __builtin_amdgcn_wmma_f32_16x16x32_bf16(false, a1, false, b0, (short)0, acc10, false, false);
    acc11 = __builtin_amdgcn_wmma_f32_16x16x32_bf16(false, a1, false, b1, (short)0, acc11, false, false);
  }
  store_tile(Cf, N16, Nreal, relu, bias, mt0,     nt0,     acc00, lane);
  store_tile(Cf, N16, Nreal, relu, bias, mt0,     nt0 + 1, acc01, lane);
  store_tile(Cf, N16, Nreal, relu, bias, mt0 + 1, nt0,     acc10, lane);
  store_tile(Cf, N16, Nreal, relu, bias, mt0 + 1, nt0 + 1, acc11, lane);
}

// ---------------------------------------------------------------------------
// Word encoder: fastText gather + char CNN (kernels 3/4/5, 50 filters each)
// xin[t] = [ ft[word_ids[t]] (300) | charCNN(t) (150) ]
// ---------------------------------------------------------------------------
__global__ void k_wordenc(const int* __restrict__ word_ids, const int* __restrict__ char_ids,
                          const float* __restrict__ ft, const float* __restrict__ cemb,
                          const float* __restrict__ cw0, const float* __restrict__ cb0,
                          const float* __restrict__ cw1, const float* __restrict__ cb1,
                          const float* __restrict__ cw2, const float* __restrict__ cb2,
                          float* __restrict__ xin) {
  int t = blockIdx.x;
  __shared__ float xs[CD_][L_];
  int tid = threadIdx.x;
  for (int i = tid; i < L_ * CD_; i += blockDim.x) {
    int pos = i / CD_, c = i % CD_;
    xs[c][pos] = cemb[(size_t)char_ids[t * L_ + pos] * CD_ + c];
  }
  __syncthreads();
  int wid = word_ids[t];
  for (int d = tid; d < D_; d += blockDim.x)
    xin[(size_t)t * XIN_ + d] = ft[(size_t)wid * D_ + d];
  if (tid < 3 * F_) {
    int ki = tid / F_, f = tid % F_;
    int kw = 3 + ki;
    const float* W  = (ki == 0 ? cw0 : ki == 1 ? cw1 : cw2) + (size_t)f * CD_ * kw;
    float bb        = (ki == 0 ? cb0 : ki == 1 ? cb1 : cb2)[f];
    float mx = -1e30f;
    for (int p = 0; p <= L_ - kw; ++p) {
      float acc = bb;
      for (int c = 0; c < CD_; ++c)
        for (int j = 0; j < kw; ++j)
          acc += W[c * kw + j] * xs[c][p + j];
      mx = fmaxf(mx, acc);
    }
    xin[(size_t)t * XIN_ + D_ + tid] = fmaxf(mx, 0.f);  // max of relu
  }
}

// ---------------------------------------------------------------------------
// BiLSTM recurrence. grid.x=2 (0=fwd, 1=bwd). Zpre = xin@Wih^T + b, [T x ZLD].
// Whh staged in LDS as bf16 (600*150*2 = 180 KB; needs CDNA5 320KB WGP LDS).
// ---------------------------------------------------------------------------
__global__ void k_lstm(const float* __restrict__ Zf, const float* __restrict__ Zb,
                       const float* __restrict__ WhhF, const float* __restrict__ WhhB,
                       float* __restrict__ wemb) {
  extern __shared__ char smem[];
  bf16_t* w = (bf16_t*)smem;                                   // [600*150]
  float*  h = (float*)(smem + (size_t)G4_ * H_ * sizeof(bf16_t));
  float*  c = h + H_;
  float*  z = c + H_;                                          // [600]
  int dir = blockIdx.x;
  const float* Zp  = dir ? Zb   : Zf;
  const float* Whh = dir ? WhhB : WhhF;
  int tid = threadIdx.x;
  for (int i = tid; i < G4_ * H_; i += blockDim.x) w[i] = (bf16_t)Whh[i];
  for (int i = tid; i < H_; i += blockDim.x) { h[i] = 0.f; c[i] = 0.f; }
  __syncthreads();
  for (int step = 0; step < T_; ++step) {
    int t = dir ? (T_ - 1 - step) : step;
    if (tid < G4_) {
      float acc = Zp[(size_t)t * ZLD_ + tid];
      const bf16_t* wr = w + (size_t)tid * H_;
      for (int k = 0; k < H_; ++k) acc += (float)wr[k] * h[k];
      z[tid] = acc;
    }
    __syncthreads();
    if (tid < H_) {
      float ig = 1.f / (1.f + __expf(-z[tid]));
      float fg = 1.f / (1.f + __expf(-z[H_ + tid]));
      float gg = tanhf(z[2 * H_ + tid]);
      float og = 1.f / (1.f + __expf(-z[3 * H_ + tid]));
      float cc = fg * c[tid] + ig * gg;
      float hh = og * tanhf(cc);
      c[tid] = cc; h[tid] = hh;
      wemb[(size_t)t * D_ + dir * H_ + tid] = hh;
    }
    __syncthreads();
  }
}

// gather windowed wemb rows into swizzled bf16 A-matrix [nsp*win x DKP]
__global__ void k_gather_wemb(const float* __restrict__ wemb, const int* __restrict__ spans,
                              int spanStride, int win, bf16_t* __restrict__ dst) {
  int row = blockIdx.x;
  int sp = row / win, j = row % win;
  int idx = spans[sp * spanStride] + j;
  idx = idx < 0 ? 0 : (idx >= T_ ? T_ - 1 : idx);
  for (int k = threadIdx.x; k < DKP_; k += blockDim.x)
    dst[asw_off(row, k, DKP_ >> 5)] =
        (bf16_t)(k < D_ ? wemb[(size_t)idx * D_ + k] : 0.f);
}

// final linear layer with N=1: out[m] = b + dot(h[m,0:K], W[:,0])
__global__ void k_lin1(const float* __restrict__ Hh, int ld, const float* __restrict__ W,
                       const float* __restrict__ b, float* __restrict__ outv, int M, int K) {
  int m = blockIdx.x * blockDim.x + threadIdx.x;
  if (m >= M) return;
  float acc = b[0];
  for (int k = 0; k < K; ++k) acc += Hh[(size_t)m * ld + k] * W[k];
  outv[m] = acc;
}

// entity span encodings: [wemb[s] | wemb[e-1] | attn(ft) | renorm(efeat)]
__global__ void k_espan(const int* __restrict__ espans, const float* __restrict__ wemb,
                        const float* __restrict__ xin, const float* __restrict__ asc,
                        const float* __restrict__ efeat, float* __restrict__ eenc) {
  int e = blockIdx.x;
  int s = espans[e * 3], en = espans[e * 3 + 1], fid = espans[e * 3 + 2];
  __shared__ float wsh[LME_];
  if (threadIdx.x == 0) {
    float mx = -1e30f;
    for (int j = 0; j < LME_; ++j) {
      float v = (j < en - s) ? asc[e * LME_ + j] : -1e9f;
      wsh[j] = v; if (v > mx) mx = v;
    }
    float ssum = 0.f;
    for (int j = 0; j < LME_; ++j) { wsh[j] = __expf(wsh[j] - mx); ssum += wsh[j]; }
    for (int j = 0; j < LME_; ++j) wsh[j] /= ssum;
  }
  __syncthreads();
  float* row = eenc + (size_t)e * SED_;
  for (int d = threadIdx.x; d < D_; d += blockDim.x) {
    row[d]       = wemb[(size_t)s * D_ + d];
    row[D_ + d]  = wemb[(size_t)(en - 1) * D_ + d];
    float acc = 0.f;
    for (int j = 0; j < LME_; ++j) {
      int idx = s + j; idx = idx < 0 ? 0 : (idx >= T_ ? T_ - 1 : idx);
      acc += wsh[j] * xin[(size_t)idx * XIN_ + d];
    }
    row[2 * D_ + d] = acc;
  }
  if (threadIdx.x < FD_) {
    const float* fv = efeat + (size_t)fid * FD_;
    float nrm = 0.f;
    for (int i = 0; i < FD_; ++i) nrm += fv[i] * fv[i];
    nrm = sqrtf(nrm);
    float sc = nrm > 1.f ? 1.f / nrm : 1.f;
    row[3 * D_ + threadIdx.x] = fv[threadIdx.x] * sc;
  }
}

// claim-span per-category encodings (grid = NC*C)
__global__ void k_cenc(const int* __restrict__ bounds, const int* __restrict__ cfeats,
                       const float* __restrict__ wemb, const float* __restrict__ xin,
                       const float* __restrict__ gsc, const float* __restrict__ cftab,
                       float* __restrict__ cenc) {
  int q = blockIdx.x / C_, cc = blockIdx.x % C_;
  int s = bounds[q * 2], en = bounds[q * 2 + 1];
  int qid = cfeats[q * C_ + cc];
  __shared__ float wsh[LMC_];
  if (threadIdx.x == 0) {
    float mx = -1e30f;
    for (int j = 0; j < LMC_; ++j) {
      float v = (j < en - s) ? gsc[(size_t)(q * LMC_ + j) * GLD_ + qid] : -1e9f;
      wsh[j] = v; if (v > mx) mx = v;
    }
    float ssum = 0.f;
    for (int j = 0; j < LMC_; ++j) { wsh[j] = __expf(wsh[j] - mx); ssum += wsh[j]; }
    for (int j = 0; j < LMC_; ++j) wsh[j] /= ssum;
  }
  __syncthreads();
  float* row = cenc + (size_t)(q * C_ + cc) * SED_;
  for (int d = threadIdx.x; d < D_; d += blockDim.x) {
    row[d]      = wemb[(size_t)s * D_ + d];
    row[D_ + d] = wemb[(size_t)(en - 1) * D_ + d];
    float acc = 0.f;
    for (int j = 0; j < LMC_; ++j) {
      int idx = s + j; idx = idx < 0 ? 0 : (idx >= T_ ? T_ - 1 : idx);
      acc += wsh[j] * xin[(size_t)idx * XIN_ + d];
    }
    row[2 * D_ + d] = acc;
  }
  if (threadIdx.x < FD_) {
    const float* fv = cftab + (size_t)qid * FD_;
    float nrm = 0.f;
    for (int i = 0; i < FD_; ++i) nrm += fv[i] * fv[i];
    nrm = sqrtf(nrm);
    float sc = nrm > 1.f ? 1.f / nrm : 1.f;
    row[3 * D_ + threadIdx.x] = fv[threadIdx.x] * sc;
  }
}

// renorm'ed actor embeddings, straight into swizzled bf16 A-matrix [NC*A x DKP]
__global__ void k_actor_renorm(const int* __restrict__ aa, const float* __restrict__ actor,
                               bf16_t* __restrict__ dst) {
  int row = blockIdx.x;  // q*A + a
  const float* av = actor + (size_t)aa[row] * D_;
  float nrm = 0.f;
  for (int d = 0; d < D_; ++d) nrm += av[d] * av[d];
  nrm = sqrtf(nrm);
  float sc = nrm > 1.f ? 1.f / nrm : 1.f;
  for (int k = threadIdx.x; k < DKP_; k += blockDim.x)
    dst[asw_off(row, k, DKP_ >> 5)] = (bf16_t)(k < D_ ? av[k] * sc : 0.f);
}

// build rel-FFNN input rows [ee | cc | ee*cc | renorm(dfeat)] in swizzled bf16
__global__ void k_pack_relin(const float* __restrict__ eenc, const float* __restrict__ cenc,
                             const float* __restrict__ dftab, const int* __restrict__ dfeatures,
                             bf16_t* __restrict__ dst) {
  int row = blockIdx.x;              // ((q*E)+e)*C + c
  int cc = row % C_; int t = row / C_;
  int e = t % E_;   int q = t / E_;
  const float* ee = eenc + (size_t)e * SED_;
  const float* ce = cenc + (size_t)(q * C_ + cc) * SED_;
  const float* dv = dftab + (size_t)dfeatures[e * NC_ + q] * FD_;
  float nrm = 0.f;
  for (int i = 0; i < FD_; ++i) nrm += dv[i] * dv[i];
  nrm = sqrtf(nrm);
  float dsc = nrm > 1.f ? 1.f / nrm : 1.f;
  for (int k = threadIdx.x; k < RELKP_; k += blockDim.x) {
    float v;
    if      (k < SED_)            v = ee[k];
    else if (k < 2 * SED_)        v = ce[k - SED_];
    else if (k < 3 * SED_)        v = ee[k - 2 * SED_] * ce[k - 2 * SED_];
    else if (k < 3 * SED_ + FD_)  v = dv[k - 3 * SED_] * dsc;
    else                          v = 0.f;
    dst[asw_off(row, k, RELKP_ >> 5)] = (bf16_t)v;
  }
}

// tmp[q,a,e,c] = link[q*A+a][e] + rel[q,e,c];  lse over e.  grid = NC*A, 128 thr
__global__ void k_final(const float* __restrict__ link, const float* __restrict__ rsc,
                        float* __restrict__ out_lse, float* __restrict__ out_tmp) {
  int q = blockIdx.x / A_, a = blockIdx.x % A_;
  int e = threadIdx.x;  // 0..127
  __shared__ float red[E_];
  float lv = link[(size_t)(q * A_ + a) * E_ + e];
  for (int cc = 0; cc < C_; ++cc) {
    float v = lv + rsc[(size_t)(q * E_ + e) * C_ + cc];
    out_tmp[(((size_t)(q * A_ + a)) * E_ + e) * C_ + cc] = v;
    red[e] = v; __syncthreads();
    for (int off = 64; off; off >>= 1) {
      if (e < off) red[e] = fmaxf(red[e], red[e + off]);
      __syncthreads();
    }
    float mx = red[0]; __syncthreads();
    red[e] = __expf(v - mx); __syncthreads();
    for (int off = 64; off; off >>= 1) {
      if (e < off) red[e] += red[e + off];
      __syncthreads();
    }
    if (e == 0) out_lse[(size_t)(q * A_ + a) * C_ + cc] = logf(red[0]) + mx;
    __syncthreads();
  }
}

// ---------------------------------------------------------------------------
// Host side
// ---------------------------------------------------------------------------
static inline size_t alignup(size_t x) { return (x + 255) & ~(size_t)255; }

static inline void launch_gemm(const bf16_t* Am, const bf16_t* Bm, const float* bias,
                               float* Cf, int M, int N, int N16, int Kp, int relu,
                               hipStream_t s) {
  int tiles  = (M / 32) * (N16 / 32);      // 32x32 macro-tiles, one per wave
  int blocks = (tiles + 7) / 8;            // 8 waves per 256-thread block
  k_gemm<<<blocks, 256, 0, s>>>(Am, Bm, bias, Cf, M, N, N16, Kp, relu);
}

extern "C" void kernel_launch(void* const* d_in, const int* in_sizes, int n_in,
                              void* d_out, int out_size, void* d_ws, size_t ws_size,
                              hipStream_t stream) {
  (void)in_sizes; (void)n_in; (void)out_size; (void)ws_size;
  char* ws = (char*)d_ws;
  size_t off = 0;
  auto alloc = [&](size_t bytes) -> void* {
    void* p = ws + off; off = alignup(off + bytes); return p;
  };

  // ---- workspace layout (~105 MB) ----
  float*  xin      = (float*) alloc((size_t)T_ * XIN_ * 4);
  bf16_t* xin_bf   = (bf16_t*)alloc((size_t)T_ * XKP_ * 2);
  bf16_t* WihF_bf  = (bf16_t*)alloc((size_t)XKP_ * ZLD_ * 2);
  bf16_t* WihB_bf  = (bf16_t*)alloc((size_t)XKP_ * ZLD_ * 2);
  float*  Zf       = (float*) alloc((size_t)T_ * ZLD_ * 4);
  float*  Zb       = (float*) alloc((size_t)T_ * ZLD_ * 4);
  float*  wemb     = (float*) alloc((size_t)T_ * D_ * 4);
  bf16_t* Aalpha   = (bf16_t*)alloc((size_t)E_ * LME_ * DKP_ * 2);
  bf16_t* Agamma   = (bf16_t*)alloc((size_t)NC_ * LMC_ * DKP_ * 2);
  bf16_t* Wa0      = (bf16_t*)alloc((size_t)DKP_ * HKP_ * 2);
  bf16_t* Wa1      = (bf16_t*)alloc((size_t)HKP_ * HKP_ * 2);
  bf16_t* Wg0      = (bf16_t*)alloc((size_t)DKP_ * HKP_ * 2);
  bf16_t* Wg1      = (bf16_t*)alloc((size_t)HKP_ * HKP_ * 2);
  bf16_t* Wg2      = (bf16_t*)alloc((size_t)HKP_ * GLD_ * 2);
  float*  h1a      = (float*) alloc((size_t)E_ * LME_ * HKP_ * 4);
  bf16_t* h1a_bf   = (bf16_t*)alloc((size_t)E_ * LME_ * HKP_ * 2);
  float*  h2a      = (float*) alloc((size_t)E_ * LME_ * HKP_ * 4);
  float*  asc      = (float*) alloc((size_t)E_ * LME_ * 4);
  float*  h1g      = (float*) alloc((size_t)NC_ * LMC_ * HKP_ * 4);
  bf16_t* h1g_bf   = (bf16_t*)alloc((size_t)NC_ * LMC_ * HKP_ * 2);
  float*  h2g      = (float*) alloc((size_t)NC_ * LMC_ * HKP_ * 4);
  bf16_t* h2g_bf   = (bf16_t*)alloc((size_t)NC_ * LMC_ * HKP_ * 2);
  float*  gsc      = (float*) alloc((size_t)NC_ * LMC_ * GLD_ * 4);
  float*  eenc     = (float*) alloc((size_t)E_ * SED_ * 4);
  bf16_t* eenc_bf  = (bf16_t*)alloc((size_t)E_ * SEDKP_ * 2);
  bf16_t* Waff     = (bf16_t*)alloc((size_t)SEDKP_ * SMLD_ * 2);
  float*  smallm   = (float*) alloc((size_t)E_ * SMLD_ * 4);
  bf16_t* smallT   = (bf16_t*)alloc((size_t)DKP_ * E_ * 2);
  bf16_t* arn_bf   = (bf16_t*)alloc((size_t)NC_ * A_ * DKP_ * 2);
  float*  linkm    = (float*) alloc((size_t)NC_ * A_ * E_ * 4);
  float*  cenc     = (float*) alloc((size_t)NC_ * C_ * SED_ * 4);
  bf16_t* relin    = (bf16_t*)alloc((size_t)NC_ * E_ * C_ * RELKP_ * 2);
  bf16_t* Wr0      = (bf16_t*)alloc((size_t)RELKP_ * HKP_ * 2);
  bf16_t* Wr1      = (bf16_t*)alloc((size_t)HKP_ * HKP_ * 2);
  float*  h1r      = (float*) alloc((size_t)NC_ * E_ * C_ * HKP_ * 4);
  bf16_t* h1r_bf   = (bf16_t*)alloc((size_t)NC_ * E_ * C_ * HKP_ * 2);
  float*  h2r      = (float*) alloc((size_t)NC_ * E_ * C_ * HKP_ * 4);
  float*  rsc      = (float*) alloc((size_t)NC_ * E_ * C_ * 4);

  auto F32 = [&](int i) { return (const float*)d_in[i]; };
  auto I32 = [&](int i) { return (const int*)d_in[i]; };

  const int PB = 256;
  auto packGrid = [&](long long n) { return (int)((n + PB - 1) / PB); };

  // ---- 1. pack weights to swizzled bf16 ----
  k_pack_wT<<<packGrid((long long)XKP_ * ZLD_), PB, 0, stream>>>(
      F32(IN_LF_WIH), XIN_, WihF_bf, XIN_, G4_, XKP_, ZLD_);
  k_pack_wT<<<packGrid((long long)XKP_ * ZLD_), PB, 0, stream>>>(
      F32(IN_LB_WIH), XIN_, WihB_bf, XIN_, G4_, XKP_, ZLD_);
  k_pack_w<<<packGrid((long long)DKP_ * HKP_), PB, 0, stream>>>(F32(IN_A0W), Wa0, D_, H_, DKP_, HKP_);
  k_pack_w<<<packGrid((long long)HKP_ * HKP_), PB, 0, stream>>>(F32(IN_A1W), Wa1, H_, H_, HKP_, HKP_);
  k_pack_w<<<packGrid((long long)DKP_ * HKP_), PB, 0, stream>>>(F32(IN_G0W), Wg0, D_, H_, DKP_, HKP_);
  k_pack_w<<<packGrid((long long)HKP_ * HKP_), PB, 0, stream>>>(F32(IN_G1W), Wg1, H_, H_, HKP_, HKP_);
  k_pack_w<<<packGrid((long long)HKP_ * GLD_), PB, 0, stream>>>(F32(IN_G2W), Wg2, H_, 30, HKP_, GLD_);
  k_pack_w<<<packGrid((long long)SEDKP_ * SMLD_), PB, 0, stream>>>(F32(IN_AFW), Waff, SED_, D_, SEDKP_, SMLD_);
  k_pack_w<<<packGrid((long long)RELKP_ * HKP_), PB, 0, stream>>>(F32(IN_R0W), Wr0, 2780, H_, RELKP_, HKP_);
  k_pack_w<<<packGrid((long long)HKP_ * HKP_), PB, 0, stream>>>(F32(IN_R1W), Wr1, H_, H_, HKP_, HKP_);

  // ---- 2. word encoder ----
  k_wordenc<<<T_, 160, 0, stream>>>(I32(IN_WORD), I32(IN_CHAR), F32(IN_FT), F32(IN_CEMB),
                                    F32(IN_CW0), F32(IN_CB0), F32(IN_CW1), F32(IN_CB1),
                                    F32(IN_CW2), F32(IN_CB2), xin);
  k_pack_a<<<packGrid((long long)T_ * XKP_), PB, 0, stream>>>(xin, XIN_, xin_bf, T_, XIN_, XKP_);

  // ---- 3. LSTM input projections (big WMMA GEMMs) ----
  launch_gemm(xin_bf, WihF_bf, F32(IN_LF_B), Zf, T_, G4_, ZLD_, XKP_, 0, stream);
  launch_gemm(xin_bf, WihB_bf, F32(IN_LB_B), Zb, T_, G4_, ZLD_, XKP_, 0, stream);

  // ---- 4. BiLSTM recurrence (Whh resident in LDS as bf16) ----
  size_t lstm_shmem = (size_t)G4_ * H_ * sizeof(bf16_t) + (size_t)(2 * H_ + G4_) * sizeof(float);
  k_lstm<<<2, 640, lstm_shmem, stream>>>(Zf, Zb, F32(IN_LF_WHH), F32(IN_LB_WHH), wemb);

  // ---- 5. gather attention windows ----
  k_gather_wemb<<<E_ * LME_, 64, 0, stream>>>(wemb, I32(IN_ESPAN), 3, LME_, Aalpha);
  k_gather_wemb<<<NC_ * LMC_, 64, 0, stream>>>(wemb, I32(IN_CB), 2, LMC_, Agamma);

  // ---- 6. alpha FFNN ----
  launch_gemm(Aalpha, Wa0, F32(IN_A0B), h1a, E_ * LME_, H_, HKP_, DKP_, 1, stream);
  k_pack_a<<<packGrid((long long)E_ * LME_ * HKP_), PB, 0, stream>>>(h1a, HKP_, h1a_bf, E_ * LME_, H_, HKP_);
  launch_gemm(h1a_bf, Wa1, F32(IN_A1B), h2a, E_ * LME_, H_, HKP_, HKP_, 1, stream);
  k_lin1<<<(E_ * LME_ + 127) / 128, 128, 0, stream>>>(h2a, HKP_, F32(IN_A2W), F32(IN_A2B),
                                                      asc, E_ * LME_, H_);

  // ---- 7. gamma FFNN ----
  launch_gemm(Agamma, Wg0, F32(IN_G0B), h1g, NC_ * LMC_, H_, HKP_, DKP_, 1, stream);
  k_pack_a<<<packGrid((long long)NC_ * LMC_ * HKP_), PB, 0, stream>>>(h1g, HKP_, h1g_bf, NC_ * LMC_, H_, HKP_);
  launch_gemm(h1g_bf, Wg1, F32(IN_G1B), h2g, NC_ * LMC_, H_, HKP_, HKP_, 1, stream);
  k_pack_a<<<packGrid((long long)NC_ * LMC_ * HKP_), PB, 0, stream>>>(h2g, HKP_, h2g_bf, NC_ * LMC_, H_, HKP_);
  launch_gemm(h2g_bf, Wg2, F32(IN_G2B), gsc, NC_ * LMC_, 30, GLD_, HKP_, 0, stream);

  // ---- 8. span encodings + affine + link ----
  k_espan<<<E_, 128, 0, stream>>>(I32(IN_ESPAN), wemb, xin, asc, F32(IN_EFEAT), eenc);
  k_pack_a<<<packGrid((long long)E_ * SEDKP_), PB, 0, stream>>>(eenc, SED_, eenc_bf, E_, SED_, SEDKP_);
  launch_gemm(eenc_bf, Waff, F32(IN_AFB), smallm, E_, D_, SMLD_, SEDKP_, 0, stream);
  k_pack_wT<<<packGrid((long long)DKP_ * E_), PB, 0, stream>>>(smallm, SMLD_, smallT, D_, E_, DKP_, E_);
  k_actor_renorm<<<NC_ * A_, 64, 0, stream>>>(I32(IN_AA), F32(IN_ACTOR), arn_bf);
  launch_gemm(arn_bf, smallT, nullptr, linkm, NC_ * A_, E_, E_, DKP_, 0, stream);

  // ---- 9. claim encodings ----
  k_cenc<<<NC_ * C_, 128, 0, stream>>>(I32(IN_CB), I32(IN_CF), wemb, xin, gsc, F32(IN_CFEAT), cenc);

  // ---- 10. relation FFNN (dominant GEMM: 4096 x 2784 x 160) ----
  k_pack_relin<<<NC_ * E_ * C_, 256, 0, stream>>>(eenc, cenc, F32(IN_DFEAT), I32(IN_DF), relin);
  launch_gemm(relin, Wr0, F32(IN_R0B), h1r, NC_ * E_ * C_, H_, HKP_, RELKP_, 1, stream);
  k_pack_a<<<packGrid((long long)NC_ * E_ * C_ * HKP_), PB, 0, stream>>>(h1r, HKP_, h1r_bf,
                                                                         NC_ * E_ * C_, H_, HKP_);
  launch_gemm(h1r_bf, Wr1, F32(IN_R1B), h2r, NC_ * E_ * C_, H_, HKP_, HKP_, 1, stream);
  k_lin1<<<(NC_ * E_ * C_ + 127) / 128, 128, 0, stream>>>(h2r, HKP_, F32(IN_R2W), F32(IN_R2B),
                                                          rsc, NC_ * E_ * C_, H_);

  // ---- 11. combine + logsumexp; out = [lse (NC*A*C) | tmp (NC*A*E*C)] ----
  float* out_lse = (float*)d_out;
  float* out_tmp = out_lse + (size_t)NC_ * A_ * C_;
  k_final<<<NC_ * A_, E_, 0, stream>>>(linkm, rsc, out_lse, out_tmp);
}
